// CrossAttention_18519898980748
// MI455X (gfx1250) — matrix operand order repaired and checked by reading, not compiled, not run on hardware
//
#include <hip/hip_runtime.h>
#include <hip/hip_bf16.h>

typedef unsigned short u16;
typedef unsigned int   u32;
typedef unsigned long long u64;
typedef __attribute__((ext_vector_type(16))) __bf16 v16bf;
typedef __attribute__((ext_vector_type(8)))  float  v8f;
typedef __attribute__((ext_vector_type(4)))  int    v4i;

#define DIM    384
#define NKEY   4096
#define NTOK   4096
#define BATCH  4
#define MTOT   (BATCH * NTOK)          // 16384 total token rows
#define KSTEPS (DIM / 32)              // 12 WMMA K-steps
#define CTILES (DIM / 16)              // 24 column tiles
#define ATT_SCALE 0.14433756729740643f // (384/8)^-0.5

#define KLD 400   // LDS stride (halves) for staged K block rows (16B aligned, conflict-spread)
#define VLD 40    // LDS stride (halves) for staged V block rows

// ---------- gfx1250 async global->LDS (guarded; falls back to load+ds_store) ----------
#if defined(__has_builtin)
#if __has_builtin(__builtin_amdgcn_global_load_async_to_lds_b128) && \
    __has_builtin(__builtin_amdgcn_s_wait_asynccnt)
#define USE_ASYNC_LDS 1
#endif
#endif
#ifndef USE_ASYNC_LDS
#define USE_ASYNC_LDS 0
#endif

__device__ __forceinline__ void cp16_g2l(const u16* g, u16* l) {
#if USE_ASYNC_LDS
  // builtin is typed on the b128 payload: (int4* global_src, int4* lds_dst, imm, imm)
  __builtin_amdgcn_global_load_async_to_lds_b128((v4i*)g, (v4i*)l, 0, 0);
#else
  *(uint4*)l = *(const uint4*)g;
#endif
}

__device__ __forceinline__ void wait_async_all() {
#if USE_ASYNC_LDS
  __builtin_amdgcn_s_wait_asynccnt(0);
#endif
}

// ---------- helpers ----------

__device__ __forceinline__ u16 f2bf_bits(float f) {
  union { __bf16 b; u16 u; } v;
  v.b = (__bf16)f;              // native v_cvt_*bf16_f32 (RTNE)
  return v.u;
}

union FragU {
  v16bf v;
  u16   us[16];
  uint4 q4[2];
};

// A-fragment (16x32, bf16) from row-major bf16 [row][k], ld in elements.
// ISA layout: row = lane&15; elems 0..7 -> K = hg*8+e ; elems 8..15 -> K = 16+hg*8+(e-8)
__device__ __forceinline__ v16bf load_a_bf16(const u16* A, int ld, int m0, int c0) {
  const int lane = threadIdx.x & 31;
  const int hg   = lane >> 4;
  const int row  = m0 + (lane & 15);
  const u16* p = A + (size_t)row * ld + c0 + hg * 8;
  FragU f;
  f.q4[0] = *(const uint4*)(p);
  f.q4[1] = *(const uint4*)(p + 16);
  return f.v;
}

// A-fragment from row-major f32, converted to bf16 on the fly.
__device__ __forceinline__ v16bf load_a_f32(const float* A, int ld, int m0, int c0) {
  const int lane = threadIdx.x & 31;
  const int hg   = lane >> 4;
  const int row  = m0 + (lane & 15);
  const float* p = A + (size_t)row * ld + c0 + hg * 8;
  FragU f;
#pragma unroll
  for (int e = 0; e < 8; ++e) f.us[e]     = f2bf_bits(p[e]);
#pragma unroll
  for (int e = 0; e < 8; ++e) f.us[8 + e] = f2bf_bits(p[16 + e]);
  return f.v;
}

// B-fragment (32x16) for logical B[k][n], sourced from k-contiguous storage S[n][k].
// ISA layout: n = lane&15; elems 0..7 -> K = k0 + hg*8 + e ; elems 8..15 -> K = k0 + 16 + hg*8 + (e-8)
__device__ __forceinline__ v16bf load_b_kmajor(const u16* S, int ld, int n0, int k0) {
  const int lane = threadIdx.x & 31;
  const int hg   = lane >> 4;
  const int n    = n0 + (lane & 15);
  const u16* p = S + (size_t)n * ld + k0 + hg * 8;
  FragU f;
  f.q4[0] = *(const uint4*)(p);
  f.q4[1] = *(const uint4*)(p + 16);
  return f.v;
}

__device__ __forceinline__ v8f wmma_bf16(v16bf a, v16bf b, v8f c) {
  return __builtin_amdgcn_wmma_f32_16x16x32_bf16(false, a, false, b, (short)0, c,
                                                 false, false);
}

// ---------- kernel 0: weights -> bf16, transposed (WT[w][d][k] = W[w][k][d]) ----------

__global__ void prep_weights_kernel(const float* Wq, const float* Wk,
                                    const float* Wv, const float* Wp, u16* WT) {
  int gid = blockIdx.x * blockDim.x + threadIdx.x;
  if (gid >= 4 * DIM * DIM) return;
  int w   = gid / (DIM * DIM);
  int rem = gid % (DIM * DIM);
  int d   = rem / DIM;   // output row (n)
  int k   = rem % DIM;   // output col (reduction dim, contiguous)
  const float* src = (w == 0) ? Wq : (w == 1) ? Wk : (w == 2) ? Wv : Wp;
  WT[gid] = f2bf_bits(src[k * DIM + d]);
}

// ---------- kernel 1: QKV projections (z = 0:q 1:k 2:v) ----------

__global__ void __launch_bounds__(256) proj_qkv_kernel(
    const float* x, const float* t, const u16* WT,
    u16* qb, u16* kb, u16* vT) {
  const int z    = blockIdx.z;
  const int wave = blockIdx.x * 8 + (threadIdx.x >> 5);
  const int mt   = wave / CTILES;
  const int nt   = wave % CTILES;
  const float* A = (z == 0) ? t : x;
  const u16*   W = WT + (size_t)z * DIM * DIM;

  v8f acc = {};
#pragma unroll
  for (int i = 0; i < KSTEPS; ++i) {
    v16bf a = load_a_f32(A, DIM, mt * 16, i * 32);
    v16bf b = load_b_kmajor(W, DIM, nt * 16, i * 32);
    acc = wmma_bf16(a, b, acc);
  }
  const int lane = threadIdx.x & 31;
  const int hg   = lane >> 4;
  const int n    = lane & 15;
#pragma unroll
  for (int r = 0; r < 8; ++r) {
    int m   = mt * 16 + hg * 8 + r;     // global token row
    int col = nt * 16 + n;
    u16 val = f2bf_bits(acc[r]);
    if (z == 0)      qb[(size_t)m * DIM + col] = val;
    else if (z == 1) kb[(size_t)m * DIM + col] = val;
    else {
      int bb  = m >> 12;                // batch
      int key = m & 4095;
      vT[((size_t)bb * DIM + col) * NKEY + key] = val;   // vT[b][c][key]
    }
  }
}

// ---------- kernel 2: flash attention ----------
// One wave = 16 query rows x full 384 out dim. All 8 waves of a block share the
// same batch + key block, so K/V blocks are staged cooperatively in
// double-buffered LDS via async global->LDS copies.

__global__ void __launch_bounds__(256) flash_kernel(
    const u16* qb, const u16* kb, const u16* vT, float* o) {
  const int tid    = threadIdx.x;
  const int wave   = blockIdx.x * 8 + (tid >> 5);
  const int lane   = tid & 31;
  const int hg     = lane >> 4;
  const int qtPerB = NTOK / 16;
  const int b      = wave / qtPerB;
  const int qt     = wave % qtPerB;

  const u16* Q  = qb + (size_t)(b * NTOK + qt * 16) * DIM;
  const u16* Kb = kb + (size_t)b * NKEY * DIM;
  const u16* Vt = vT + (size_t)b * DIM * NKEY;

  // staged K block: 32 rows x 384 halves; staged V block: 384 rows x 32 halves
  __shared__ __align__(16) u16 Ksh[2][32 * KLD];    // 51200 B
  __shared__ __align__(16) u16 Vsh[2][DIM * VLD];   // 61440 B
  __shared__ __align__(16) u16 pS[8][16 * 40];      // 10240 B
  u16* myP = pS[tid >> 5];

  // cooperative stage of one 32-key block into LDS buffer `buf`
  auto stage_kv = [&](int key0, int buf) {
    u16* Kdst = Ksh[buf];
    u16* Vdst = Vsh[buf];
#pragma unroll
    for (int j = 0; j < 6; ++j) {           // K: 32 rows x 48 16B-chunks = 1536
      int ch  = tid + j * 256;
      int row = ch / 48, seg = ch % 48;
      cp16_g2l(Kb + (size_t)(key0 + row) * DIM + seg * 8,
               Kdst + row * KLD + seg * 8);
    }
#pragma unroll
    for (int j = 0; j < 6; ++j) {           // V: 384 rows x 4 16B-chunks = 1536
      int ch  = tid + j * 256;
      int row = ch / 4, seg = ch % 4;
      cp16_g2l(Vt + (size_t)row * NKEY + key0 + seg * 8,
               Vdst + row * VLD + seg * 8);
    }
  };

  v16bf qf[KSTEPS];
#pragma unroll
  for (int i = 0; i < KSTEPS; ++i) qf[i] = load_a_bf16(Q, DIM, 0, i * 32);

  v8f acc[CTILES];
  const v8f vzero = {0.f, 0.f, 0.f, 0.f, 0.f, 0.f, 0.f, 0.f};
#pragma unroll
  for (int tt = 0; tt < CTILES; ++tt) acc[tt] = vzero;

  float mrow[8], lrow[8];
#pragma unroll
  for (int r = 0; r < 8; ++r) { mrow[r] = -3.0e38f; lrow[r] = 0.f; }

  stage_kv(0, 0);

  for (int key0 = 0; key0 < NKEY; key0 += 32) {
    const int cur = (key0 >> 5) & 1;
    wait_async_all();          // own async copies into Ksh/Vsh[cur] complete
    __syncthreads();           // everyone's copies visible; prev buffer free
    if (key0 + 32 < NKEY) stage_kv(key0 + 32, cur ^ 1);

    const u16* Kcur = Ksh[cur];
    const u16* Vcur = Vsh[cur];

    // S = Q * K^T for 32 keys (two 16x16 D-tiles), B-frags from LDS
    v8f s0 = vzero, s1 = vzero;
#pragma unroll
    for (int i = 0; i < KSTEPS; ++i) {
      v16bf bk0 = load_b_kmajor(Kcur, KLD, 0,  i * 32);
      v16bf bk1 = load_b_kmajor(Kcur, KLD, 16, i * 32);
      s0 = wmma_bf16(qf[i], bk0, s0);
      s1 = wmma_bf16(qf[i], bk1, s1);
    }

    // online softmax per row (row = hg*8 + r), reductions within 16-lane group
#pragma unroll
    for (int r = 0; r < 8; ++r) {
      float a = s0[r] * ATT_SCALE;
      float c = s1[r] * ATT_SCALE;
      float mx = fmaxf(a, c);
#pragma unroll
      for (int off = 8; off >= 1; off >>= 1)
        mx = fmaxf(mx, __shfl_xor(mx, off, 32));
      float mnew = fmaxf(mrow[r], mx);
      float corr = __expf(mrow[r] - mnew);
      float p0   = __expf(a - mnew);
      float p1   = __expf(c - mnew);
      mrow[r] = mnew;
      float ps = p0 + p1;
#pragma unroll
      for (int off = 8; off >= 1; off >>= 1)
        ps += __shfl_xor(ps, off, 32);
      lrow[r] = lrow[r] * corr + ps;
#pragma unroll
      for (int tt = 0; tt < CTILES; ++tt) acc[tt][r] *= corr;
      const int row = hg * 8 + r;
      myP[row * 40 + (lane & 15)]      = f2bf_bits(p0);
      myP[row * 40 + 16 + (lane & 15)] = f2bf_bits(p1);
    }

    // O += P * V  (P from LDS in A-fragment layout, V-frags from staged LDS)
    v16bf pf = load_a_bf16(myP, 40, 0, 0);
#pragma unroll
    for (int tt = 0; tt < CTILES; ++tt) {
      v16bf vf = load_b_kmajor(Vcur, VLD, tt * 16, 0);
      acc[tt] = wmma_bf16(pf, vf, acc[tt]);
    }
  }

  // epilogue: normalize and store f32 o[b][token][c]
  const int n = lane & 15;
#pragma unroll
  for (int tt = 0; tt < CTILES; ++tt) {
#pragma unroll
    for (int r = 0; r < 8; ++r) {
      int row = qt * 16 + hg * 8 + r;
      o[((size_t)b * NTOK + row) * DIM + tt * 16 + n] = acc[tt][r] / lrow[r];
    }
  }
}

// ---------- kernel 3: faithful transpose(0,2,1).reshape permutation, f32 -> bf16 ----------

__global__ void permute_kernel(const float* o, u16* operm) {
  int gid = blockIdx.x * blockDim.x + threadIdx.x;
  const int per = NTOK * DIM;
  if (gid >= BATCH * per) return;
  int b  = gid / per;
  int i  = gid % per;           // flat index into per-batch (C, TN) view
  int r  = i >> 12;             // i / 4096  -> channel of o
  int nn = i & 4095;            // i % 4096  -> token of o
  operm[gid] = f2bf_bits(o[(size_t)b * per + (size_t)nn * DIM + r]);
}

// ---------- kernel 4: output projection + bias ----------

__global__ void __launch_bounds__(256) out_proj_kernel(
    const u16* operm, const u16* WTp, const float* bp, float* out) {
  const int wave = blockIdx.x * 8 + (threadIdx.x >> 5);
  const int mt   = wave / CTILES;
  const int nt   = wave % CTILES;
  v8f acc = {};
#pragma unroll
  for (int i = 0; i < KSTEPS; ++i) {
    v16bf a = load_a_bf16(operm, DIM, mt * 16, i * 32);
    v16bf b = load_b_kmajor(WTp, DIM, nt * 16, i * 32);
    acc = wmma_bf16(a, b, acc);
  }
  const int lane = threadIdx.x & 31;
  const int hg   = lane >> 4;
  const int n    = lane & 15;
#pragma unroll
  for (int r = 0; r < 8; ++r) {
    int m   = mt * 16 + hg * 8 + r;
    int col = nt * 16 + n;
    out[(size_t)m * DIM + col] = acc[r] + bp[col];
  }
}

// ---------- launch ----------

extern "C" void kernel_launch(void* const* d_in, const int* in_sizes, int n_in,
                              void* d_out, int out_size, void* d_ws, size_t ws_size,
                              hipStream_t stream) {
  const float* x  = (const float*)d_in[0];
  const float* t  = (const float*)d_in[1];
  const float* Wq = (const float*)d_in[2];
  const float* Wk = (const float*)d_in[3];
  const float* Wv = (const float*)d_in[4];
  const float* Wp = (const float*)d_in[5];
  const float* bp = (const float*)d_in[6];
  float* out = (float*)d_out;

  char* ws = (char*)d_ws;
  size_t off = 0;
  auto alloc = [&](size_t bytes) {
    void* p = ws + off;
    off = (off + bytes + 255) & ~(size_t)255;
    return p;
  };
  u16*   WT    = (u16*)  alloc((size_t)4 * DIM * DIM * sizeof(u16));
  u16*   qb    = (u16*)  alloc((size_t)MTOT * DIM * sizeof(u16));
  u16*   kb    = (u16*)  alloc((size_t)MTOT * DIM * sizeof(u16));
  u16*   vT    = (u16*)  alloc((size_t)MTOT * DIM * sizeof(u16));
  float* obuf  = (float*)alloc((size_t)MTOT * DIM * sizeof(float));
  u16*   operm = (u16*)  alloc((size_t)MTOT * DIM * sizeof(u16));
  (void)ws_size; (void)in_sizes; (void)n_in; (void)out_size;

  prep_weights_kernel<<<(4 * DIM * DIM + 255) / 256, 256, 0, stream>>>(
      Wq, Wk, Wv, Wp, WT);

  dim3 gproj((MTOT / 16) * CTILES / 8, 1, 3);  // 3072 blocks x 8 waves = tiles
  proj_qkv_kernel<<<gproj, 256, 0, stream>>>(x, t, WT, qb, kb, vT);

  flash_kernel<<<(MTOT / 16) / 8, 256, 0, stream>>>(qb, kb, vT, obuf);  // 128 blocks

  permute_kernel<<<(MTOT * DIM + 255) / 256, 256, 0, stream>>>(obuf, operm);

  out_proj_kernel<<<(MTOT / 16) * CTILES / 8, 256, 0, stream>>>(
      operm, WT + (size_t)3 * DIM * DIM, bp, out);
}
